// XlaQuantizedLinear_63445256896863
// MI455X (gfx1250) — compile-verified, matrix operand order checked by reading, not run on hardware
//
#include <hip/hip_runtime.h>
#include <hip/hip_bf16.h>

// out[M,N] = (x[M,K] @ w[N,K]^T) * scale[N]
// x fp32, w int8-in-int32, scale fp32, out fp32.
// M=8192, K=4096, N=11008 -> all divisible by (BM=128, BN=128, BK=32).

typedef __attribute__((ext_vector_type(16))) _Float16 v16h;
typedef __attribute__((ext_vector_type(8)))  _Float16 v8h;
typedef __attribute__((ext_vector_type(8)))  float    v8f;
typedef __attribute__((ext_vector_type(4)))  float    f32x4;
typedef __attribute__((ext_vector_type(4)))  int      i32x4;

#define BM 128
#define BN 128
#define BK 32
#define LDSS (BK + 8)   // 40 halves = 80B row stride: conflict-free b128 row fan

__global__ __launch_bounds__(256)
void XlaQuantizedLinear_wmma_f16(const float* __restrict__ x,
                                 const int*   __restrict__ w,
                                 const float* __restrict__ wscale,
                                 float*       __restrict__ out,
                                 int M, int N, int K)
{
    __shared__ _Float16 shA[2][BM * LDSS];
    __shared__ _Float16 shB[2][BN * LDSS];

    const int tid  = threadIdx.x;
    const int lane = tid & 31;
    const int wave = tid >> 5;

    const int bm = blockIdx.x * BM;
    const int bn = blockIdx.y * BN;

    // 8 waves: 2 in M x 4 in N; each wave computes 64(M) x 32(N)
    const int wm = (wave & 1) * 64;
    const int wn = (wave >> 1) * 32;

    // WMMA f16 A/B fragment addressing (ISA 7.12.2):
    // lane L holds row (L&15), K = (L>=16 ? 8 : 0) + {0..7} and same +16.
    const int fr   = lane & 15;
    const int koff = (lane >> 4) * 8;           // in halves

    // Cooperative loaders: 2 threads per row, 16 elements each
    const int lrow = tid >> 1;                  // 0..127
    const int lcol = (tid & 1) * 16;            // 0 or 16

    const float* aptr = x + (size_t)(bm + lrow) * K + lcol;
    const int*   bptr = w + (size_t)(bn + lrow) * K + lcol;

    v8f acc[4][2];
#pragma unroll
    for (int mi = 0; mi < 4; ++mi)
#pragma unroll
        for (int ni = 0; ni < 2; ++ni)
            acc[mi][ni] = (v8f){0.f, 0.f, 0.f, 0.f, 0.f, 0.f, 0.f, 0.f};

    f32x4 areg[4];
    i32x4 breg[4];

    auto ld_global = [&](int k0) {
#pragma unroll
        for (int i = 0; i < 4; ++i) {
            areg[i] = *(const f32x4*)(aptr + k0 + i * 4);
            breg[i] = *(const i32x4*)(bptr + k0 + i * 4);
        }
        // Pull tile k0+BK toward L2/WGP$ while this one is consumed.
        if (k0 + BK < K) {
            __builtin_prefetch(aptr + k0 + BK, 0, 1);
            __builtin_prefetch(bptr + k0 + BK, 0, 1);
        }
    };

    auto st_lds = [&](int buf) {
        v8h a0, a1, b0, b1;
#pragma unroll
        for (int i = 0; i < 4; ++i) {
#pragma unroll
            for (int j = 0; j < 4; ++j) {
                const int e = i * 4 + j;
                const _Float16 av = (_Float16)areg[i][j];
                const _Float16 bv = (_Float16)(float)breg[i][j];
                if (e < 8) { a0[e] = av; b0[e] = bv; }
                else       { a1[e - 8] = av; b1[e - 8] = bv; }
            }
        }
        _Float16* da = &shA[buf][lrow * LDSS + lcol];
        _Float16* db = &shB[buf][lrow * LDSS + lcol];
        *(v8h*)(da)     = a0;
        *(v8h*)(da + 8) = a1;
        *(v8h*)(db)     = b0;
        *(v8h*)(db + 8) = b1;
    };

    auto compute = [&](int buf) {
        v16h afrag[4], bfrag[2];
#pragma unroll
        for (int mi = 0; mi < 4; ++mi) {
            const _Float16* p = &shA[buf][(wm + mi * 16 + fr) * LDSS + koff];
            v8h lo = *(const v8h*)(p);
            v8h hi = *(const v8h*)(p + 16);
            afrag[mi] = __builtin_shufflevector(lo, hi,
                0, 1, 2, 3, 4, 5, 6, 7, 8, 9, 10, 11, 12, 13, 14, 15);
        }
#pragma unroll
        for (int ni = 0; ni < 2; ++ni) {
            const _Float16* p = &shB[buf][(wn + ni * 16 + fr) * LDSS + koff];
            v8h lo = *(const v8h*)(p);
            v8h hi = *(const v8h*)(p + 16);
            bfrag[ni] = __builtin_shufflevector(lo, hi,
                0, 1, 2, 3, 4, 5, 6, 7, 8, 9, 10, 11, 12, 13, 14, 15);
        }
#pragma unroll
        for (int mi = 0; mi < 4; ++mi)
#pragma unroll
            for (int ni = 0; ni < 2; ++ni)
                acc[mi][ni] = __builtin_amdgcn_wmma_f32_16x16x32_f16(
                    /*neg_a=*/false, afrag[mi],
                    /*neg_b=*/false, bfrag[ni],
                    /*c_mod=*/(short)0, acc[mi][ni],
                    /*reuse_a=*/false, /*reuse_b=*/false);
    };

    const int nk = K / BK;

    // Prologue: stage k-step 0 into buffer 0.
    ld_global(0);
    st_lds(0);
    __syncthreads();

    for (int kt = 0; kt < nk; ++kt) {
        const int cur = kt & 1;
        if (kt + 1 < nk) ld_global((kt + 1) * BK);  // in flight behind WMMAs
        compute(cur);
        if (kt + 1 < nk) {
            __syncthreads();          // everyone done reading buffer cur^1
            st_lds(cur ^ 1);
            __syncthreads();          // buffer cur^1 visible for next step
        }
    }

    // Epilogue: D layout -> lane L, VGPR v holds M = v + (L>=16 ? 8:0), N = L&15.
    const int mhalf = (lane >> 4) * 8;
#pragma unroll
    for (int ni = 0; ni < 2; ++ni) {
        const int n  = bn + wn + ni * 16 + fr;
        const float sc = wscale[n];
#pragma unroll
        for (int mi = 0; mi < 4; ++mi) {
            const int m0 = bm + wm + mi * 16 + mhalf;
#pragma unroll
            for (int v = 0; v < 8; ++v)
                out[(size_t)(m0 + v) * N + n] = acc[mi][ni][v] * sc;
        }
    }
}

extern "C" void kernel_launch(void* const* d_in, const int* in_sizes, int n_in,
                              void* d_out, int out_size, void* d_ws, size_t ws_size,
                              hipStream_t stream) {
    const float* x  = (const float*)d_in[0];
    const int*   w  = (const int*)d_in[1];
    const float* sc = (const float*)d_in[2];
    float* out = (float*)d_out;

    const int N = in_sizes[2];              // OUT = 11008
    const int K = in_sizes[1] / N;          // IN  = 4096
    const int M = in_sizes[0] / K;          // B*S = 8192

    dim3 grid(M / BM, N / BN);              // (64, 86)
    XlaQuantizedLinear_wmma_f16<<<grid, 256, 0, stream>>>(x, w, sc, out, M, N, K);
}
